// MultiHeadAttention_44263932952657
// MI455X (gfx1250) — matrix operand order repaired and checked by reading, not compiled, not run on hardware
//
#include <hip/hip_runtime.h>
#include <hip/hip_bf16.h>

typedef _Float16 h16 __attribute__((ext_vector_type(16)));
typedef _Float16 h8v __attribute__((ext_vector_type(8)));
typedef float    f8  __attribute__((ext_vector_type(8)));

#define S_LEN 2048
#define DMODEL 1024
#define NHEAD 16
#define HDIM 64
#define MROWS 4096   // B*S

__device__ __forceinline__ f8 f8zero() {
    f8 z;
#pragma unroll
    for (int i = 0; i < 8; ++i) z[i] = 0.0f;
    return z;
}

__device__ __forceinline__ f8 wmma16(h16 a, h16 b, f8 c) {
    // D = A(16x32 f16) x B(32x16 f16) + C(16x16 f32)
    return __builtin_amdgcn_wmma_f32_16x16x32_f16(false, a, false, b, (short)0, c, false, false);
}

// A fragment (16x32 f16): lane m = lane&15, hh = lane>>4.
// elements 0..7  -> k = k0 + 8*hh + (0..7)      (contiguous)
// elements 8..15 -> k = k0 + 16 + 8*hh + (0..7) (contiguous)
__device__ __forceinline__ h16 load_a_frag(const _Float16* rowptr, int k0, int hh) {
    const _Float16* p = rowptr + k0 + 8 * hh;
    h8v lo = *(const h8v*)p;
    h8v hi = *(const h8v*)(p + 16);
    h16 a;
#pragma unroll
    for (int i = 0; i < 8; ++i) { a[i] = lo[i]; a[i + 8] = hi[i]; }
    return a;
}

// B fragment (32x16 f16): lane n = lane&15, hh = lane>>4; element e<8 -> k = k0 + 8*hh + e, e>=8 -> k = k0 + 16 + 8*hh + (e-8).
__device__ __forceinline__ h16 load_b_frag(const _Float16* rowptr, int k0, int hh) {
    const _Float16* p = rowptr + k0 + 8 * hh;
    h8v lo = *(const h8v*)p;
    h8v hi = *(const h8v*)(p + 16);
    h16 b;
#pragma unroll
    for (int i = 0; i < 8; ++i) { b[i] = lo[i]; b[i + 8] = hi[i]; }
    return b;
}

__device__ __forceinline__ float rmax16(float v) {
#pragma unroll
    for (int off = 1; off <= 8; off <<= 1) v = fmaxf(v, __shfl_xor(v, off, 32));
    return v;
}
__device__ __forceinline__ float rsum16(float v) {
#pragma unroll
    for (int off = 1; off <= 8; off <<= 1) v += __shfl_xor(v, off, 32);
    return v;
}

// ---------------- converts ----------------
__global__ void cvt_f16(const float* __restrict__ in, _Float16* __restrict__ out, int n) {
    int i = blockIdx.x * 256 + threadIdx.x;
    if (i < n) out[i] = (_Float16)in[i];
}
// out[c][r] = in[r][c]; grid = (cols/256, rows), no div/mod
__global__ void cvtT_f16(const float* __restrict__ in, _Float16* __restrict__ out,
                         int rows, int cols) {
    const int c = blockIdx.x * 256 + threadIdx.x;
    const int r = blockIdx.y;
    out[(size_t)c * rows + r] = (_Float16)in[(size_t)r * cols + c];
}

// ------- shared GEMM core: 4 waves x (32M x 64N), K=1024, ping-pong pipelined -------
// acc[0..3] = rows m..m+15, acc[4..7] = rows m+16..m+31 (within wave tile)
__device__ __forceinline__ void gemm_core(const _Float16* __restrict__ A,
                                          const _Float16* __restrict__ WT,
                                          int mb, int nb, int w, int nn, int hh,
                                          f8 acc[8]) {
    const _Float16* ar0 = A + (size_t)(mb + w * 32 + nn) * DMODEL;
    const _Float16* ar1 = ar0 + 16 * DMODEL;
    const _Float16* br0 = WT + (size_t)(nb + nn) * DMODEL;
    const _Float16* br1 = br0 + 16 * DMODEL;
    const _Float16* br2 = br0 + 32 * DMODEL;
    const _Float16* br3 = br0 + 48 * DMODEL;

    // stage X preloaded at k=0; each stage reloads IN PLACE (no rotation copies)
    h16 xa0 = load_a_frag(ar0, 0, hh), xa1 = load_a_frag(ar1, 0, hh);
    h16 xb0 = load_b_frag(br0, 0, hh), xb1 = load_b_frag(br1, 0, hh);
    h16 xb2 = load_b_frag(br2, 0, hh), xb3 = load_b_frag(br3, 0, hh);

    for (int k0 = 0; k0 < DMODEL - 64; k0 += 64) {
        h16 ya0 = load_a_frag(ar0, k0 + 32, hh);
        h16 ya1 = load_a_frag(ar1, k0 + 32, hh);
        h16 yb0 = load_b_frag(br0, k0 + 32, hh);
        h16 yb1 = load_b_frag(br1, k0 + 32, hh);
        h16 yb2 = load_b_frag(br2, k0 + 32, hh);
        h16 yb3 = load_b_frag(br3, k0 + 32, hh);
        acc[0] = wmma16(xa0, xb0, acc[0]);
        acc[1] = wmma16(xa0, xb1, acc[1]);
        acc[2] = wmma16(xa0, xb2, acc[2]);
        acc[3] = wmma16(xa0, xb3, acc[3]);
        acc[4] = wmma16(xa1, xb0, acc[4]);
        acc[5] = wmma16(xa1, xb1, acc[5]);
        acc[6] = wmma16(xa1, xb2, acc[6]);
        acc[7] = wmma16(xa1, xb3, acc[7]);
        xa0 = load_a_frag(ar0, k0 + 64, hh);
        xa1 = load_a_frag(ar1, k0 + 64, hh);
        xb0 = load_b_frag(br0, k0 + 64, hh);
        xb1 = load_b_frag(br1, k0 + 64, hh);
        xb2 = load_b_frag(br2, k0 + 64, hh);
        xb3 = load_b_frag(br3, k0 + 64, hh);
        acc[0] = wmma16(ya0, yb0, acc[0]);
        acc[1] = wmma16(ya0, yb1, acc[1]);
        acc[2] = wmma16(ya0, yb2, acc[2]);
        acc[3] = wmma16(ya0, yb3, acc[3]);
        acc[4] = wmma16(ya1, yb0, acc[4]);
        acc[5] = wmma16(ya1, yb1, acc[5]);
        acc[6] = wmma16(ya1, yb2, acc[6]);
        acc[7] = wmma16(ya1, yb3, acc[7]);
    }
    // tail: x holds k=960; load y at k=992
    {
        h16 ya0 = load_a_frag(ar0, DMODEL - 32, hh);
        h16 ya1 = load_a_frag(ar1, DMODEL - 32, hh);
        h16 yb0 = load_b_frag(br0, DMODEL - 32, hh);
        h16 yb1 = load_b_frag(br1, DMODEL - 32, hh);
        h16 yb2 = load_b_frag(br2, DMODEL - 32, hh);
        h16 yb3 = load_b_frag(br3, DMODEL - 32, hh);
        acc[0] = wmma16(xa0, xb0, acc[0]);
        acc[1] = wmma16(xa0, xb1, acc[1]);
        acc[2] = wmma16(xa0, xb2, acc[2]);
        acc[3] = wmma16(xa0, xb3, acc[3]);
        acc[4] = wmma16(xa1, xb0, acc[4]);
        acc[5] = wmma16(xa1, xb1, acc[5]);
        acc[6] = wmma16(xa1, xb2, acc[6]);
        acc[7] = wmma16(xa1, xb3, acc[7]);
        acc[0] = wmma16(ya0, yb0, acc[0]);
        acc[1] = wmma16(ya0, yb1, acc[1]);
        acc[2] = wmma16(ya0, yb2, acc[2]);
        acc[3] = wmma16(ya0, yb3, acc[3]);
        acc[4] = wmma16(ya1, yb0, acc[4]);
        acc[5] = wmma16(ya1, yb1, acc[5]);
        acc[6] = wmma16(ya1, yb2, acc[6]);
        acc[7] = wmma16(ya1, yb3, acc[7]);
    }
}

// ---------------- fused QKV GEMM + head split ----------------
__global__ __launch_bounds__(128) void gemm_qkv(const _Float16* __restrict__ Xh,
                                                const _Float16* __restrict__ WT,
                                                const float* __restrict__ bias,
                                                _Float16* __restrict__ Qm,
                                                _Float16* __restrict__ Km,
                                                _Float16* __restrict__ VTm) {
    const int lane = threadIdx.x & 31, w = threadIdx.x >> 5;
    const int nn = lane & 15, hh = lane >> 4;
    const int mb = blockIdx.y * 128, nb = blockIdx.x * 64;

    f8 acc[8];
#pragma unroll
    for (int i = 0; i < 8; ++i) acc[i] = f8zero();
    gemm_core(Xh, WT, mb, nb, w, nn, hh, acc);

#pragma unroll
    for (int nt = 0; nt < 4; ++nt) {
        const int j = nb + nt * 16 + nn;         // 0..3071
        const float bv = bias[j];
        const int t = j >> 10;                    // 0=Q 1=K 2=V
        const int hd = (j & 1023) >> 6;           // head
        const int d = j & 63;
#pragma unroll
        for (int g = 0; g < 2; ++g) {
#pragma unroll
            for (int r = 0; r < 8; ++r) {
                const int m = mb + w * 32 + g * 16 + r + 8 * hh;  // [0,4096)
                const int bb = m >> 11, s = m & 2047;
                const _Float16 hv = (_Float16)(acc[g * 4 + nt][r] + bv);
                if (t == 0)
                    Qm[((size_t)(bb * NHEAD + hd) * S_LEN + s) * HDIM + d] = hv;
                else if (t == 1)
                    Km[((size_t)(bb * NHEAD + hd) * S_LEN + s) * HDIM + d] = hv;
                else
                    VTm[((size_t)(bb * NHEAD + hd) * HDIM + d) * S_LEN + s] = hv;
            }
        }
    }
}

// ---------------- flash attention (causal) ----------------
// grid: (S/64 q-tiles, B*H). block: 128 (4 waves, 16 q-rows each).
__global__ __launch_bounds__(128) void attn(const _Float16* __restrict__ Qm,
                                            const _Float16* __restrict__ Km,
                                            const _Float16* __restrict__ VTm,
                                            _Float16* __restrict__ Ctx) {
    __shared__ __align__(16) _Float16 Pbuf[4][16][80];  // per-wave, padded rows (160B)
    const int lane = threadIdx.x & 31, w = threadIdx.x >> 5;
    const int nn = lane & 15, hh = lane >> 4;
    const int qt = blockIdx.x, bh = blockIdx.y;
    const int qb = qt * 64;
    const _Float16* Qbh = Qm + (size_t)bh * S_LEN * HDIM;
    const _Float16* Kbh = Km + (size_t)bh * S_LEN * HDIM;
    const _Float16* Vbh = VTm + (size_t)bh * HDIM * S_LEN;

    // preload Q fragments, fold in softmax scale 1/sqrt(64)=0.125 (exact in f16)
    const _Float16* qrowp = Qbh + (size_t)(qb + w * 16 + nn) * HDIM;
    h16 aQ0 = load_a_frag(qrowp, 0, hh);
    h16 aQ1 = load_a_frag(qrowp, 32, hh);
    const _Float16 qs = (_Float16)0.125f;
#pragma unroll
    for (int i = 0; i < 16; ++i) { aQ0[i] *= qs; aQ1[i] *= qs; }

    float mI[8], lI[8];
    f8 Oacc[4];
#pragma unroll
    for (int r = 0; r < 8; ++r) { mI[r] = -1e30f; lI[r] = 0.0f; }
#pragma unroll
    for (int dt = 0; dt < 4; ++dt) Oacc[dt] = f8zero();

    // K fragments for tile 0 preloaded; refilled in place each iteration
    h16 bK[4][2];
#pragma unroll
    for (int nt = 0; nt < 4; ++nt) {
        const _Float16* krow = Kbh + (size_t)(nt * 16 + nn) * HDIM;
        bK[nt][0] = load_b_frag(krow, 0, hh);
        bK[nt][1] = load_b_frag(krow, 32, hh);
    }

    for (int kt = 0; kt <= qt; ++kt) {
        const int kb = kt * 64;

        // --- scores: 8 WMMAs on resident/prefetched K fragments ---
        f8 S[4];
#pragma unroll
        for (int nt = 0; nt < 4; ++nt) S[nt] = f8zero();
#pragma unroll
        for (int nt = 0; nt < 4; ++nt) {
            S[nt] = wmma16(aQ0, bK[nt][0], S[nt]);
            S[nt] = wmma16(aQ1, bK[nt][1], S[nt]);
        }

        // --- issue all V fragment loads now; softmax math below hides them ---
        h16 bV[2][4];
#pragma unroll
        for (int ks = 0; ks < 2; ++ks)
#pragma unroll
            for (int dt = 0; dt < 4; ++dt)
                bV[ks][dt] = load_b_frag(Vbh + (size_t)(dt * 16 + nn) * S_LEN + kb,
                                         ks * 32, hh);

        const bool diag = (kt == qt);
        if (diag) {
#pragma unroll
            for (int nt = 0; nt < 4; ++nt) {
                const int key = kb + nt * 16 + nn;
#pragma unroll
                for (int r = 0; r < 8; ++r) {
                    const int qg = qb + w * 16 + r + 8 * hh;
                    S[nt][r] = (key <= qg) ? S[nt][r] : -1e30f;
                }
            }
        }

        // online softmax row stats (row m=r+8*hh lives across the 16-lane half-group)
#pragma unroll
        for (int r = 0; r < 8; ++r) {
            float v = fmaxf(fmaxf(S[0][r], S[1][r]), fmaxf(S[2][r], S[3][r]));
            v = rmax16(v);
            const float mnew = fmaxf(mI[r], v);
            const float alpha = __expf(mI[r] - mnew);
            mI[r] = mnew;
            lI[r] *= alpha;
#pragma unroll
            for (int dt = 0; dt < 4; ++dt) Oacc[dt][r] *= alpha;
        }
#pragma unroll
        for (int r = 0; r < 8; ++r) {
            float ps = 0.0f;
#pragma unroll
            for (int nt = 0; nt < 4; ++nt) {
                const float p = __expf(S[nt][r] - mI[r]);
                S[nt][r] = p;
                ps += p;
            }
            lI[r] += rsum16(ps);
        }

        // C-layout P -> LDS (f16), then reread as A-layout fragments
#pragma unroll
        for (int nt = 0; nt < 4; ++nt)
#pragma unroll
            for (int r = 0; r < 8; ++r)
                Pbuf[w][r + 8 * hh][nt * 16 + nn] = (_Float16)S[nt][r];

        // --- prefetch next tile's K fragments in place (consumed next iteration) ---
        if (kt < qt) {
            const int kb2 = kb + 64;
#pragma unroll
            for (int nt = 0; nt < 4; ++nt) {
                const _Float16* krow = Kbh + (size_t)(kb2 + nt * 16 + nn) * HDIM;
                bK[nt][0] = load_b_frag(krow, 0, hh);
                bK[nt][1] = load_b_frag(krow, 32, hh);
            }
        }
        __syncthreads();

        // O += P @ V  (V fragments already in flight / resident)
#pragma unroll
        for (int ks = 0; ks < 2; ++ks) {
            h16 aP = load_a_frag(&Pbuf[w][nn][0], ks * 32, hh);
#pragma unroll
            for (int dt = 0; dt < 4; ++dt)
                Oacc[dt] = wmma16(aP, bV[ks][dt], Oacc[dt]);
        }
        __syncthreads();
    }

    // finalize, write context [B,S,D] f16 at column head*64+d
    const int bb = bh >> 4, head = bh & 15;
#pragma unroll
    for (int r = 0; r < 8; ++r) {
        const float inv = 1.0f / lI[r];
        const int s = qb + w * 16 + r + 8 * hh;
#pragma unroll
        for (int dt = 0; dt < 4; ++dt) {
            Ctx[((size_t)(bb * S_LEN + s)) * DMODEL + head * HDIM + dt * 16 + nn] =
                (_Float16)(Oacc[dt][r] * inv);
        }
    }
}

// ---------------- output projection ----------------
__global__ __launch_bounds__(128) void gemm_out(const _Float16* __restrict__ Ctx,
                                                const _Float16* __restrict__ WT,
                                                const float* __restrict__ bias,
                                                float* __restrict__ out) {
    const int lane = threadIdx.x & 31, w = threadIdx.x >> 5;
    const int nn = lane & 15, hh = lane >> 4;
    const int mb = blockIdx.y * 128, nb = blockIdx.x * 64;

    f8 acc[8];
#pragma unroll
    for (int i = 0; i < 8; ++i) acc[i] = f8zero();
    gemm_core(Ctx, WT, mb, nb, w, nn, hh, acc);

#pragma unroll
    for (int nt = 0; nt < 4; ++nt) {
        const int j = nb + nt * 16 + nn;
        const float bv = bias[j];
#pragma unroll
        for (int g = 0; g < 2; ++g) {
#pragma unroll
            for (int r = 0; r < 8; ++r) {
                const int m = mb + w * 32 + g * 16 + r + 8 * hh;
                out[(size_t)m * DMODEL + j] = acc[g * 4 + nt][r] + bv;
            }
        }
    }
}

extern "C" void kernel_launch(void* const* d_in, const int* in_sizes, int n_in,
                              void* d_out, int out_size, void* d_ws, size_t ws_size,
                              hipStream_t stream) {
    (void)in_sizes; (void)n_in; (void)out_size; (void)ws_size;
    const float* x     = (const float*)d_in[0];  // [2,2048,1024]
    const float* W_qkv = (const float*)d_in[1];  // [1024,3072]
    const float* b_qkv = (const float*)d_in[2];  // [3072]
    const float* W_out = (const float*)d_in[3];  // [1024,1024]
    const float* b_out = (const float*)d_in[4];  // [1024]

    char* p = (char*)d_ws;
    _Float16* Xh    = (_Float16*)p; p += (size_t)MROWS * DMODEL * 2;      // 8 MB
    _Float16* WqkvT = (_Float16*)p; p += (size_t)3 * DMODEL * DMODEL * 2; // 6 MB
    _Float16* WoutT = (_Float16*)p; p += (size_t)DMODEL * DMODEL * 2;     // 2 MB
    _Float16* Qm    = (_Float16*)p; p += (size_t)MROWS * DMODEL * 2;      // 8 MB
    _Float16* Km    = (_Float16*)p; p += (size_t)MROWS * DMODEL * 2;      // 8 MB
    _Float16* VTm   = (_Float16*)p; p += (size_t)MROWS * DMODEL * 2;      // 8 MB
    _Float16* Ctx   = (_Float16*)p; p += (size_t)MROWS * DMODEL * 2;      // 8 MB

    const int nx = MROWS * DMODEL;
    cvt_f16<<<(nx + 255) / 256, 256, 0, stream>>>(x, Xh, nx);
    cvtT_f16<<<dim3(3 * DMODEL / 256, DMODEL), 256, 0, stream>>>(W_qkv, WqkvT, DMODEL, 3 * DMODEL);
    cvtT_f16<<<dim3(DMODEL / 256, DMODEL), 256, 0, stream>>>(W_out, WoutT, DMODEL, DMODEL);

    gemm_qkv<<<dim3(3 * DMODEL / 64, MROWS / 128), 128, 0, stream>>>(Xh, WqkvT, b_qkv, Qm, Km, VTm);
    attn<<<dim3(S_LEN / 64, 2 * NHEAD), 128, 0, stream>>>(Qm, Km, VTm, Ctx);
    gemm_out<<<dim3(DMODEL / 64, MROWS / 128), 128, 0, stream>>>(Ctx, WoutT, b_out, (float*)d_out);
}
